// Skyformer_14989435863282
// MI455X (gfx1250) — compile-verified, matrix operand order checked by reading, not run on hardware
//
#include <hip/hip_runtime.h>

#define B_ 4
#define H_ 8
#define N_ 2048
#define N2_ 4096
#define D_ 64
#define F_ 128
#define MACC_ 4
#define DNORM 0.42044820762685725f   // 32^(-0.25)

typedef __attribute__((ext_vector_type(16))) _Float16 v16h;
typedef __attribute__((ext_vector_type(8)))  float    v8f;
typedef __attribute__((ext_vector_type(2)))  float    v2f;

// Inverse of the 16-bit 16xK fragment layout (CDNA5 ISA §7.12.2):
// lane = m16 | (hiK<<4);  elem e: p=e>>1,o=e&1; p<4 -> k=8*hiK+2p+o ; p>=4 -> k=16+8*hiK+2(p-4)+o
__device__ __forceinline__ void f16_frag_pos(int m16, int kk, int& lane, int& e) {
    int hi16 = kk >> 4;
    int rem  = kk & 15;
    int hiK  = rem >> 3;
    int w    = rem & 7;
    int p    = (hi16 ? 4 : 0) + (w >> 1);
    int o    = w & 1;
    lane = m16 | (hiK << 4);
    e    = 2 * p + o;
}

// ---------------- Kernel 1: scale/mask X=concat(q,k), pack A-fragments (f16), diag1 ----
__global__ void prep_pack_x(const float* __restrict__ q, const float* __restrict__ k,
                            const float* __restrict__ mask,
                            _Float16* __restrict__ xpack, float* __restrict__ diag1) {
    int idx = blockIdx.x * blockDim.x + threadIdx.x;
    if (idx >= B_ * H_ * N2_) return;
    int bh = idx / N2_;
    int r  = idx - bh * N2_;
    int b  = bh / H_;
    int rm = (r < N_) ? r : (r - N_);
    const float* src = ((r < N_) ? q : k) + ((size_t)bh * N_ + rm) * D_;
    float mv = mask[b * N_ + rm] * DNORM;
    float x[D_];
    float s = 0.f;
    for (int j = 0; j < D_; ++j) { float vv = src[j] * mv; x[j] = vv; s += vv * vv; }
    diag1[idx] = 0.5f * s;
    int mtile = r >> 4, m16 = r & 15;
    size_t base = (((size_t)bh * 256 + mtile) * 2) * 32 * 16;
    for (int kk = 0; kk < D_; ++kk) {
        int kt = kk >> 5, kin = kk & 31, lane, e;
        f16_frag_pos(m16, kin, lane, e);
        xpack[base + ((size_t)kt * 32 + lane) * 16 + e] = (_Float16)x[kk];
    }
}

// ---------------- Kernel 2: gather sketch rows, pack B-fragments (f16), diag2 ----------
__global__ void gather_pack_xs(const float* __restrict__ q, const float* __restrict__ k,
                               const float* __restrict__ mask, const int* __restrict__ sket,
                               _Float16* __restrict__ xspack, float* __restrict__ diag2) {
    int idx = blockIdx.x * blockDim.x + threadIdx.x;
    if (idx >= B_ * H_ * MACC_ * F_) return;
    int f  = idx & 127;
    int m  = (idx >> 7) & 3;
    int hh = (idx >> 9) & 7;
    int b  = idx >> 12;
    int bh = b * H_ + hh;
    int srow = sket[(b * MACC_ + m) * F_ + f];
    int rm = (srow < N_) ? srow : (srow - N_);
    const float* src = ((srow < N_) ? q : k) + ((size_t)bh * N_ + rm) * D_;
    float mv = mask[b * N_ + rm] * DNORM;
    float x[D_];
    float s = 0.f;
    for (int j = 0; j < D_; ++j) { float vv = src[j] * mv; x[j] = vv; s += vv * vv; }
    diag2[((size_t)bh * MACC_ + m) * F_ + f] = 0.5f * s;
    int nt = f >> 4, n16 = f & 15;
    size_t base = ((((size_t)bh * MACC_ + m) * 8 + nt) * 2) * 32 * 16;
    for (int kk = 0; kk < D_; ++kk) {
        int kt = kk >> 5, kin = kk & 31, lane, e;
        f16_frag_pos(n16, kin, lane, e);
        xspack[base + ((size_t)kt * 32 + lane) * 16 + e] = (_Float16)x[kk];
    }
}

// ---------------- Kernel 3: AS = sum_m sign * exp(X @ XS[m]^T - d1 - d2)  (f16 WMMA) ---
__global__ void __launch_bounds__(256)
as_wmma(const _Float16* __restrict__ xpack, const _Float16* __restrict__ xspack,
        const float* __restrict__ diag1, const float* __restrict__ diag2,
        const float* __restrict__ rsign, float* __restrict__ AS) {
    int bh   = blockIdx.x >> 5;     // 32 row-blocks of 128 per (b,h)
    int mblk = blockIdx.x & 31;
    int b    = bh / H_;
    int wave = threadIdx.x >> 5, lane = threadIdx.x & 31;
    int mtile = mblk * 8 + wave;    // 16-row tile index (0..255)
    int row0  = mtile * 16;
    int m16 = lane & 15, hi = lane >> 4, Mhi = hi * 8;
    float d1[8];
#pragma unroll
    for (int r = 0; r < 8; ++r) d1[r] = diag1[(size_t)bh * N2_ + row0 + Mhi + r];
    const _Float16* ax = xpack + (((size_t)bh * 256 + mtile) * 2) * 32 * 16;
    for (int nt = 0; nt < 8; ++nt) {
        int col = nt * 16 + m16;
        v8f acc = {};
#pragma unroll
        for (int m = 0; m < MACC_; ++m) {
            v8f c = {};
            const _Float16* bx = xspack + ((((size_t)bh * MACC_ + m) * 8 + nt) * 2) * 32 * 16;
#pragma unroll
            for (int kt = 0; kt < 2; ++kt) {
                v16h a  = *(const v16h*)(ax + ((size_t)kt * 32 + lane) * 16);
                v16h bb = *(const v16h*)(bx + ((size_t)kt * 32 + lane) * 16);
                c = __builtin_amdgcn_wmma_f32_16x16x32_f16(false, a, false, bb,
                                                           (short)0, c, false, false);
            }
            float d2 = diag2[((size_t)bh * MACC_ + m) * F_ + col];
            float sg = rsign[((size_t)b * MACC_ + m) * F_ + col];
#pragma unroll
            for (int r = 0; r < 8; ++r)
                acc[r] += sg * __expf(c[r] - d1[r] - d2);
        }
#pragma unroll
        for (int r = 0; r < 8; ++r)
            AS[((size_t)bh * N2_ + row0 + Mhi + r) * F_ + col] = acc[r];
    }
}

// ---------------- Kernel 4: STAS gather + 0.1*I ---------------------------------------
__global__ void stas_gather(const float* __restrict__ AS, const int* __restrict__ sket,
                            const float* __restrict__ rsign, float* __restrict__ STAS) {
    int idx = blockIdx.x * blockDim.x + threadIdx.x;
    if (idx >= B_ * H_ * F_ * F_) return;
    int e  = idx & 127;
    int dd = (idx >> 7) & 127;
    int bh = idx >> 14;
    int b  = bh / H_;
    float s = 0.f;
#pragma unroll
    for (int m = 0; m < MACC_; ++m) {
        int srow = sket[(b * MACC_ + m) * F_ + dd];
        float sg = rsign[((size_t)b * MACC_ + m) * F_ + dd];
        s += sg * AS[((size_t)bh * N2_ + srow) * F_ + e];
    }
    if (dd == e) s += 0.1f;
    STAS[idx] = s;
}

// ---------------- Kernel 5: D_inv = rsqrt(rowsum) -------------------------------------
__global__ void row_dinv(const float* __restrict__ STAS, float* __restrict__ Dinv) {
    int idx = blockIdx.x * blockDim.x + threadIdx.x;
    if (idx >= B_ * H_ * F_) return;
    const float* row = STAS + (size_t)idx * F_;
    float s = 0.f;
    for (int e = 0; e < F_; ++e) s += row[e];
    Dinv[idx] = rsqrtf(s);
}

// ---------------- Kernel 5b: STAS_n = Dinv[d]*STAS*Dinv[e] (in place) -----------------
__global__ void stas_normalize(float* __restrict__ STAS, const float* __restrict__ Dinv) {
    int idx = blockIdx.x * blockDim.x + threadIdx.x;
    if (idx >= B_ * H_ * F_ * F_) return;
    int e  = idx & 127;
    int dd = (idx >> 7) & 127;
    int bh = idx >> 14;
    STAS[idx] = Dinv[bh * F_ + dd] * STAS[idx] * Dinv[bh * F_ + e];
}

// ---------------- Kernel 6a/6b/6c: V0 = (1/global_max_colsum) * K^T -------------------
__global__ void colmax_per_bh(const float* __restrict__ K, float* __restrict__ bhmax) {
    __shared__ float cs[128];
    int bh = blockIdx.x, t = threadIdx.x;          // 128 threads
    const float* Kb = K + (size_t)bh * F_ * F_;
    float s = 0.f;
    for (int dd = 0; dd < F_; ++dd) s += Kb[dd * F_ + t];
    cs[t] = s;
    __syncthreads();
    for (int off = 64; off > 0; off >>= 1) {
        if (t < off) cs[t] = fmaxf(cs[t], cs[t + off]);
        __syncthreads();
    }
    if (t == 0) bhmax[bh] = cs[0];
}

__global__ void reduce_max32(const float* __restrict__ bhmax, float* __restrict__ gmax) {
    float v = bhmax[threadIdx.x];                  // 32 threads = one wave
    for (int off = 16; off > 0; off >>= 1) v = fmaxf(v, __shfl_down(v, off, 32));
    if (threadIdx.x == 0) gmax[0] = v;
}

__global__ void init_v(const float* __restrict__ K, const float* __restrict__ gmax,
                       float* __restrict__ V) {
    int bh = blockIdx.x, t = threadIdx.x;          // 128 threads
    float inv = 1.0f / gmax[0];
    const float* Kb = K + (size_t)bh * F_ * F_;
    float* Vb = V + (size_t)bh * F_ * F_;
    for (int dd = 0; dd < F_; ++dd) Vb[dd * F_ + t] = inv * Kb[t * F_ + dd];
}

// ---------------- Kernel 7: C = scale * A @ (useDiag ? diagCoef*I - B : B)  (f32 WMMA) -
__global__ void __launch_bounds__(256)
mm128(const float* __restrict__ A, const float* __restrict__ Bm, float* __restrict__ C,
      float diagCoef, int useDiag, float scale) {
    int bh = blockIdx.x;
    const float* Ab = A  + (size_t)bh * F_ * F_;
    const float* Bb = Bm + (size_t)bh * F_ * F_;
    float* Cb       = C  + (size_t)bh * F_ * F_;
    int wave = threadIdx.x >> 5, lane = threadIdx.x & 31;
    int m16 = lane & 15, hi = lane >> 4, Mhi = hi * 8;
    int row0 = wave * 16;
    for (int nt = 0; nt < 8; ++nt) {
        int col = nt * 16 + m16;
        v8f c = {};
        for (int ks = 0; ks < 32; ++ks) {
            int kb = ks * 4 + hi * 2;
            v2f a  = { Ab[(size_t)(row0 + m16) * F_ + kb],
                       Ab[(size_t)(row0 + m16) * F_ + kb + 1] };
            v2f bv = { Bb[(size_t)kb * F_ + col],
                       Bb[(size_t)(kb + 1) * F_ + col] };
            if (useDiag) {
                bv[0] = diagCoef * ((kb     == col) ? 1.f : 0.f) - bv[0];
                bv[1] = diagCoef * ((kb + 1 == col) ? 1.f : 0.f) - bv[1];
            }
            c = __builtin_amdgcn_wmma_f32_16x16x4_f32(false, a, false, bv,
                                                      (short)0, c, false, false);
        }
#pragma unroll
        for (int r = 0; r < 8; ++r)
            Cb[(size_t)(row0 + Mhi + r) * F_ + col] = scale * c[r];
    }
}

// ---------------- Kernel 8: Kfin = ((K_AS*Dinv) @ STAS_inv) * Dinv * mask -------------
__global__ void __launch_bounds__(256)
k_transform(const float* __restrict__ AS, const float* __restrict__ Vinv,
            const float* __restrict__ Dinv, const float* __restrict__ mask,
            float* __restrict__ Kfin) {
    int bh   = blockIdx.x >> 4;
    int mblk = blockIdx.x & 15;
    int b    = bh / H_;
    const float* Ab = AS + (size_t)bh * N2_ * F_ + (size_t)N_ * F_;   // K half of AS
    const float* Bb = Vinv + (size_t)bh * F_ * F_;
    const float* dv = Dinv + (size_t)bh * F_;
    float* Cb = Kfin + (size_t)bh * N_ * F_;
    int wave = threadIdx.x >> 5, lane = threadIdx.x & 31;
    int m16 = lane & 15, hi = lane >> 4, Mhi = hi * 8;
    int row0 = mblk * 128 + wave * 16;
    for (int nt = 0; nt < 8; ++nt) {
        int col = nt * 16 + m16;
        float dcol = dv[col];
        v8f c = {};
        for (int ks = 0; ks < 32; ++ks) {
            int kb = ks * 4 + hi * 2;
            v2f a  = { Ab[(size_t)(row0 + m16) * F_ + kb]     * dv[kb],
                       Ab[(size_t)(row0 + m16) * F_ + kb + 1] * dv[kb + 1] };
            v2f bv = { Bb[(size_t)kb * F_ + col],
                       Bb[(size_t)(kb + 1) * F_ + col] };
            c = __builtin_amdgcn_wmma_f32_16x16x4_f32(false, a, false, bv,
                                                      (short)0, c, false, false);
        }
#pragma unroll
        for (int r = 0; r < 8; ++r) {
            int rr = row0 + Mhi + r;
            Cb[(size_t)rr * F_ + col] = c[r] * dcol * mask[b * N_ + rr];
        }
    }
}

// ---------------- Kernel 9: context = Kfin^T @ v  (128x64, K=2048) --------------------
__global__ void __launch_bounds__(256)
context_wmma(const float* __restrict__ Kfin, const float* __restrict__ vsrc,
             float* __restrict__ ctx) {
    int bh = blockIdx.x;
    const float* Ab = Kfin + (size_t)bh * N_ * F_;
    const float* Vb = vsrc + (size_t)bh * N_ * D_;
    float* Cb = ctx + (size_t)bh * F_ * D_;
    int wave = threadIdx.x >> 5, lane = threadIdx.x & 31;
    int m16 = lane & 15, hi = lane >> 4, Mhi = hi * 8;
    int f0 = wave * 16;
    for (int et = 0; et < 4; ++et) {
        int col = et * 16 + m16;
        v8f c = {};
        for (int ks = 0; ks < N_ / 4; ++ks) {
            int kb = ks * 4 + hi * 2;
            v2f a  = { Ab[(size_t)kb * F_ + f0 + m16],          // transposed A access
                       Ab[(size_t)(kb + 1) * F_ + f0 + m16] };
            v2f bv = { Vb[(size_t)kb * D_ + col],
                       Vb[(size_t)(kb + 1) * D_ + col] };
            c = __builtin_amdgcn_wmma_f32_16x16x4_f32(false, a, false, bv,
                                                      (short)0, c, false, false);
        }
#pragma unroll
        for (int r = 0; r < 8; ++r)
            Cb[(size_t)(f0 + Mhi + r) * D_ + col] = c[r];
    }
}

// ---------------- Kernel 10: out = Q_AS @ context  (2048x64, K=128) -------------------
__global__ void __launch_bounds__(256)
final_wmma(const float* __restrict__ AS, const float* __restrict__ ctx,
           float* __restrict__ out) {
    int bh   = blockIdx.x >> 4;
    int mblk = blockIdx.x & 15;
    const float* Ab = AS + (size_t)bh * N2_ * F_;                 // Q half of AS
    const float* Bb = ctx + (size_t)bh * F_ * D_;
    float* Ob = out + (size_t)bh * N_ * D_;
    int wave = threadIdx.x >> 5, lane = threadIdx.x & 31;
    int m16 = lane & 15, hi = lane >> 4, Mhi = hi * 8;
    int row0 = mblk * 128 + wave * 16;
    for (int et = 0; et < 4; ++et) {
        int col = et * 16 + m16;
        v8f c = {};
        for (int ks = 0; ks < 32; ++ks) {
            int kb = ks * 4 + hi * 2;
            v2f a  = { Ab[(size_t)(row0 + m16) * F_ + kb],
                       Ab[(size_t)(row0 + m16) * F_ + kb + 1] };
            v2f bv = { Bb[(size_t)kb * D_ + col],
                       Bb[(size_t)(kb + 1) * D_ + col] };
            c = __builtin_amdgcn_wmma_f32_16x16x4_f32(false, a, false, bv,
                                                      (short)0, c, false, false);
        }
#pragma unroll
        for (int r = 0; r < 8; ++r)
            Ob[(size_t)(row0 + Mhi + r) * D_ + col] = c[r];
    }
}

// ---------------- Host orchestration ---------------------------------------------------
extern "C" void kernel_launch(void* const* d_in, const int* in_sizes, int n_in,
                              void* d_out, int out_size, void* d_ws, size_t ws_size,
                              hipStream_t stream) {
    (void)in_sizes; (void)n_in; (void)out_size; (void)ws_size;
    const float* q     = (const float*)d_in[0];
    const float* k     = (const float*)d_in[1];
    const float* v     = (const float*)d_in[2];
    const float* mask  = (const float*)d_in[3];
    const float* rsign = (const float*)d_in[4];
    const int*   sket  = (const int*)d_in[5];

    char* ws = (char*)d_ws;
    _Float16* xpack  = (_Float16*)(ws + 0);           // 16,777,216 B
    _Float16* xspack = (_Float16*)(ws + 16777216);    //  2,097,152 B
    float* diag1 = (float*)(ws + 18874368);           //    524,288 B
    float* diag2 = (float*)(ws + 19398656);           //     65,536 B
    float* AS    = (float*)(ws + 19464192);           // 67,108,864 B
    float* STAS  = (float*)(ws + 86573056);           //  2,097,152 B
    float* Dinv  = (float*)(ws + 88670208);           //     16,384 B
    float* V     = (float*)(ws + 88686592);           //  2,097,152 B
    float* V2    = (float*)(ws + 90783744);           //  2,097,152 B
    float* KV    = (float*)(ws + 92880896);           //  2,097,152 B
    float* T     = (float*)(ws + 94978048);           //  2,097,152 B
    float* U     = (float*)(ws + 97075200);           //  2,097,152 B
    float* Kfin  = (float*)(ws + 99172352);           // 33,554,432 B
    float* ctx   = (float*)(ws + 132726784);          //  1,048,576 B
    float* bhmax = (float*)(ws + 133775360);          //        128 B
    float* gmax  = (float*)(ws + 133775488);          //          4 B
    float* out   = (float*)d_out;

    prep_pack_x   <<<512, 256, 0, stream>>>(q, k, mask, xpack, diag1);
    gather_pack_xs<<<64,  256, 0, stream>>>(q, k, mask, sket, xspack, diag2);
    as_wmma       <<<1024, 256, 0, stream>>>(xpack, xspack, diag1, diag2, rsign, AS);
    stas_gather   <<<2048, 256, 0, stream>>>(AS, sket, rsign, STAS);
    row_dinv      <<<16,  256, 0, stream>>>(STAS, Dinv);
    stas_normalize<<<2048, 256, 0, stream>>>(STAS, Dinv);
    colmax_per_bh <<<32, 128, 0, stream>>>(STAS, bhmax);
    reduce_max32  <<<1, 32, 0, stream>>>(bhmax, gmax);
    init_v        <<<32, 128, 0, stream>>>(STAS, gmax, V);

    float* vc = V; float* vn = V2;
    for (int it = 0; it < 6; ++it) {
        mm128<<<32, 256, 0, stream>>>(STAS, vc, KV, 0.f,  0, 1.f);   // KV = K@V
        mm128<<<32, 256, 0, stream>>>(KV,   KV, T,  7.f,  1, 1.f);   // T  = KV@(7I-KV)
        mm128<<<32, 256, 0, stream>>>(KV,   T,  U,  15.f, 1, 1.f);   // U  = KV@(15I-T)
        mm128<<<32, 256, 0, stream>>>(vc,   U,  vn, 13.f, 1, 0.25f); // V' = 0.25*V@(13I-U)
        float* tmp = vc; vc = vn; vn = tmp;
    }

    k_transform <<<512, 256, 0, stream>>>(AS, vc, Dinv, mask, Kfin);
    context_wmma<<<32,  256, 0, stream>>>(Kfin, v, ctx);
    final_wmma  <<<512, 256, 0, stream>>>(AS, ctx, out);
}
